// GraphFeatureNet_60627758350799
// MI455X (gfx1250) — matrix operand order repaired and checked by reading, not compiled, not run on hardware
//
#include <hip/hip_runtime.h>
#include <hip/hip_bf16.h>

typedef __attribute__((ext_vector_type(16))) __bf16 v16bf;
typedef __attribute__((ext_vector_type(8)))  __bf16 v8bf;
typedef __attribute__((ext_vector_type(8)))  float  v8f;

namespace {
constexpr int Bn = 8;     // batch
constexpr int Nn = 768;   // nodes
constexpr int Tn = 769;   // tokens = N+1
constexpr int TP = 832;   // padded tokens (13*64)
constexpr int Dn = 256;   // model dim
constexpr int Hn = 8;     // heads
constexpr int DKn = 32;   // head dim
constexpr int Fn = 128;   // input feature dim
constexpr int Ln = 6;     // layers
constexpr int Dm = 256;   // mlp hidden
constexpr int MP = Bn * TP; // 6656 padded rows total
}

// ---------------------------------------------------------------------------
// Generic WMMA GEMM: C[M,N] = act(A[M,K] @ B + bias) (+ residual)
//   BT=false: B is (K,N) row-major.   BT=true: B is (N,K) row-major (B^T GEMM).
// Tile: 64(M) x NT(N) per workgroup, 8 waves.
//   NT=64: each wave computes 16x32 (2 accumulators, shared A fragment).
//   NT=32: each wave computes 16x16 (1 accumulator) -- used for attn@V (N=32).
// blockIdx.z advances A/B/C by zA/zB/zC elements (per-head addressing).
// ---------------------------------------------------------------------------
template<int NT, bool BT, bool BIAS, bool GELU, bool RES>
__global__ __launch_bounds__(256)
void gemm_ker(const float* __restrict__ A, int lda, long long zA,
              const float* __restrict__ Bm, int ldb, long long zB,
              const float* __restrict__ bias,
              const float* __restrict__ res,
              float* __restrict__ C, int ldc, long long zC,
              int K)
{
    static_assert(NT == 32 || NT == 64, "NT must be 32 or 64");
    constexpr int NACC = NT / 32;   // accumulators per wave

    A  += (long long)blockIdx.z * zA;
    Bm += (long long)blockIdx.z * zB;
    C  += (long long)blockIdx.z * zC;
    if (RES) res += (long long)blockIdx.z * zC;

    const int m0 = blockIdx.y * 64;
    const int n0 = blockIdx.x * NT;

    __shared__ alignas(32) __bf16 As[64][32];  // [m][k]
    __shared__ alignas(32) __bf16 Bs[NT][32];  // [n][k] (transposed stage)

    const int tid   = threadIdx.x;
    const int w     = tid >> 5;
    const int lane  = tid & 31;
    const int msub  = (w >> 1) << 4;                                // 0,16,32,48
    const int nbase = (NT == 64) ? ((w & 1) << 5) : ((w & 1) << 4); // 0/32 or 0/16
    const int lm    = lane & 15;
    const int lh    = lane >> 4;

    v8f acc[NACC] = {};

    for (int k0 = 0; k0 < K; k0 += 32) {
        __syncthreads();
        {   // stage A tile 64x32 (8 contiguous elems/thread)
            const int r = tid >> 2;
            const int c = (tid & 3) << 3;
            const float* src = A + (long long)(m0 + r) * lda + (k0 + c);
            __bf16* dst = &As[r][c];
            #pragma unroll
            for (int i = 0; i < 8; ++i) dst[i] = (__bf16)src[i];
            if (k0 + 32 < K) __builtin_prefetch(src + 32, 0, 1);
        }
        if (BT) {   // Bs[n][k] = Bm[(n0+n)*ldb + k0+k]  (row copies)
            if (NT == 64) {
                const int r = tid >> 2;
                const int c = (tid & 3) << 3;
                const float* src = Bm + (long long)(n0 + r) * ldb + (k0 + c);
                __bf16* dst = &Bs[r][c];
                #pragma unroll
                for (int i = 0; i < 8; ++i) dst[i] = (__bf16)src[i];
                if (k0 + 32 < K) __builtin_prefetch(src + 32, 0, 1);
            } else {
                const int n = tid >> 3;
                const int c = (tid & 7) << 2;
                const float* src = Bm + (long long)(n0 + n) * ldb + (k0 + c);
                #pragma unroll
                for (int i = 0; i < 4; ++i) Bs[n][c + i] = (__bf16)src[i];
            }
        } else {    // Bs[n][k] = Bm[(k0+k)*ldb + n0+n]  (transpose stage)
            if (NT == 64) {
                const int kk = tid >> 3;
                const int n  = (tid & 7) << 3;
                const float* src = Bm + (long long)(k0 + kk) * ldb + (n0 + n);
                #pragma unroll
                for (int i = 0; i < 8; ++i) Bs[n + i][kk] = (__bf16)src[i];
                if (k0 + 32 < K) __builtin_prefetch(src + (long long)32 * ldb, 0, 1);
            } else {
                const int kk = tid >> 3;
                const int n  = (tid & 7) << 2;
                const float* src = Bm + (long long)(k0 + kk) * ldb + (n0 + n);
                #pragma unroll
                for (int i = 0; i < 4; ++i) Bs[n + i][kk] = (__bf16)src[i];
            }
        }
        __syncthreads();

        // A fragment: lane half 0 -> K 0..15 ; half 1 -> K 16..31
        union { v16bf v; v8bf h[2]; } af;
        const __bf16* ar = &As[msub + lm][0];
        af.v = *(const v16bf*)(ar + (lh << 4));
        // B fragments: lane half 0 -> K 0..15 ; half 1 -> K 16..31 (col n = lm)
        #pragma unroll
        for (int s = 0; s < NACC; ++s) {
            union { v16bf v; v8bf h[2]; } bfr;
            bfr.v = *(const v16bf*)(&Bs[nbase + (s << 4) + lm][lh << 4]);
            acc[s] = __builtin_amdgcn_wmma_f32_16x16x32_bf16(false, af.v, false, bfr.v,
                                                             (short)0, acc[s], false, false);
        }
    }

    #pragma unroll
    for (int s = 0; s < NACC; ++s) {
        const int col = n0 + nbase + (s << 4) + lm;
        float bv = 0.0f;
        if (BIAS) bv = bias[col];
        #pragma unroll
        for (int r = 0; r < 8; ++r) {
            const int row = m0 + msub + (lh << 3) + r;
            const long long idx = (long long)row * ldc + col;
            float v = acc[s][r] + bv;
            if (GELU) v = 0.5f * v * (1.0f + erff(v * 0.7071067811865475f));
            if (RES) v += res[idx];
            C[idx] = v;
        }
    }
}

// ---------------------------------------------------------------------------
// degree: deg[b,n] = sum_j adj[b,n,j] ; one wave per row
// ---------------------------------------------------------------------------
__global__ void deg_ker(const int* __restrict__ adj, float* __restrict__ deg)
{
    const int w = threadIdx.x >> 5, l = threadIdx.x & 31;
    const int row = blockIdx.x * 8 + w;            // [0, B*N)
    const int* ar = adj + (long long)row * Nn;
    float s = 0.f;
    for (int j = l; j < Nn; j += 32) s += (float)ar[j];
    #pragma unroll
    for (int o = 16; o; o >>= 1) s += __shfl_xor(s, o, 32);
    if (l == 0) deg[row] = s;
}

__global__ void degmax_ker(const float* __restrict__ deg, float* __restrict__ dmx)
{
    const int b = blockIdx.x;
    const float* dr = deg + b * Nn;
    float m = 0.f;
    for (int j = threadIdx.x; j < Nn; j += 256) m = fmaxf(m, dr[j]);
    __shared__ float red[8];
    #pragma unroll
    for (int o = 16; o; o >>= 1) m = fmaxf(m, __shfl_xor(m, o, 32));
    if ((threadIdx.x & 31) == 0) red[threadIdx.x >> 5] = m;
    __syncthreads();
    if (threadIdx.x == 0) {
        float mm = red[0];
        for (int i = 1; i < 8; ++i) mm = fmaxf(mm, red[i]);
        dmx[b] = mm;
    }
}

// ---------------------------------------------------------------------------
// attention bias (B,T,T): edge-type dot, -gamma*dist, no-edge bias, degree term,
// virtual-node row/col/self replacement. One block per (b,i) row.
// ---------------------------------------------------------------------------
__global__ void bias_ker(const int* __restrict__ adj, const float* __restrict__ dist,
                         const float* __restrict__ eoh, const float* __restrict__ etw,
                         const float* __restrict__ p_noedge, const float* __restrict__ p_degs,
                         const float* __restrict__ p_gamma, const float* __restrict__ p_v2n,
                         const float* __restrict__ p_n2v, const float* __restrict__ p_vself,
                         const float* __restrict__ deg, const float* __restrict__ dmx,
                         float* __restrict__ bias)
{
    const int rid = blockIdx.x;
    const int b = rid / Tn;
    const int i = rid - b * Tn;
    float* br = bias + ((long long)b * Tn + i) * Tn;

    if (i == 0) {
        const float v2n = *p_v2n, vself = *p_vself;
        for (int j = threadIdx.x; j < Tn; j += 256) br[j] = (j == 0) ? vself : v2n;
        return;
    }
    const float n2v = *p_n2v;
    const float noedge = *p_noedge, degs = *p_degs, gamma = *p_gamma;
    const float e0 = etw[0], e1 = etw[1], e2 = etw[2], e3 = etw[3];
    const int ii = i - 1;
    const float di  = deg[b * Nn + ii];
    const float dmv = dmx[b] + 1e-6f;
    const int*   arow = adj  + ((long long)b * Nn + ii) * Nn;
    const float* drow = dist + ((long long)b * Nn + ii) * Nn;
    const float* erow = eoh  + ((long long)b * Nn + ii) * Nn * 4;
    for (int j = threadIdx.x; j < Tn; j += 256) {
        if (j == 0) { br[0] = n2v; continue; }
        const int jj = j - 1;
        const float* ev = erow + (long long)jj * 4;
        float e = ev[0]*e0 + ev[1]*e1 + ev[2]*e2 + ev[3]*e3 - gamma * drow[jj];
        if (arow[jj] <= 0) e += noedge;
        e += degs * (di + deg[b * Nn + jj]) / dmv;
        br[j] = e;
    }
}

// ---------------------------------------------------------------------------
// x init: cls row at t=0, zeros in per-batch pad rows [Tn, TP)
// grid = B*64 (k=0 -> cls, k=1..63 -> pad rows 769..831), block = D
// ---------------------------------------------------------------------------
__global__ void init_ker(const float* __restrict__ cls, float* __restrict__ x)
{
    const int rid = blockIdx.x;
    const int b = rid >> 6;
    const int k = rid & 63;
    const int t = (k == 0) ? 0 : (Tn - 1 + k);
    float* row = x + ((long long)b * TP + t) * Dn;
    row[threadIdx.x] = (k == 0) ? cls[threadIdx.x] : 0.0f;
}

// ---------------------------------------------------------------------------
// LayerNorm over D=256, one block (256 threads) per real (b,t) row
// ---------------------------------------------------------------------------
__global__ void ln_ker(const float* __restrict__ x, const float* __restrict__ g,
                       const float* __restrict__ bt, float* __restrict__ y)
{
    const int rid = blockIdx.x;
    const int b = rid / Tn;
    const int t = rid - b * Tn;
    const long long off = ((long long)b * TP + t) * Dn + threadIdx.x;
    const float v = x[off];
    float s1 = v, s2 = v * v;
    #pragma unroll
    for (int o = 16; o; o >>= 1) { s1 += __shfl_xor(s1, o, 32); s2 += __shfl_xor(s2, o, 32); }
    __shared__ float r1[8], r2[8];
    __shared__ float mean_s, rstd_s;
    if ((threadIdx.x & 31) == 0) { r1[threadIdx.x >> 5] = s1; r2[threadIdx.x >> 5] = s2; }
    __syncthreads();
    if (threadIdx.x == 0) {
        float a = 0.f, q = 0.f;
        for (int i = 0; i < 8; ++i) { a += r1[i]; q += r2[i]; }
        const float mean = a * (1.0f / Dn);
        const float var  = q * (1.0f / Dn) - mean * mean;
        mean_s = mean;
        rstd_s = rsqrtf(var + 1e-5f);
    }
    __syncthreads();
    y[off] = (v - mean_s) * rstd_s * g[threadIdx.x] + bt[threadIdx.x];
}

// ---------------------------------------------------------------------------
// softmax over keys: row = scores[h, i, :]; val = row*1/sqrt(DK) + bias[b,i,:]
// pads (j >= Tn) written as exact 0. grid = (Tn, Hn), block = 256. In-place.
// ---------------------------------------------------------------------------
__global__ void softmax_ker(float* __restrict__ sc, const float* __restrict__ bias, int b)
{
    const int i = blockIdx.x, h = blockIdx.y;
    float* row = sc + ((long long)h * TP + i) * TP;
    const float* br = bias + ((long long)b * Tn + i) * Tn;
    const float scale = 0.17677669529663687f;  // 1/sqrt(32)
    const int tid = threadIdx.x;
    __shared__ float red[8];
    __shared__ float bval;

    float mx = -1e30f;
    for (int j = tid; j < Tn; j += 256) mx = fmaxf(mx, row[j] * scale + br[j]);
    #pragma unroll
    for (int o = 16; o; o >>= 1) mx = fmaxf(mx, __shfl_xor(mx, o, 32));
    if ((tid & 31) == 0) red[tid >> 5] = mx;
    __syncthreads();
    if (tid == 0) {
        float m = red[0];
        for (int k = 1; k < 8; ++k) m = fmaxf(m, red[k]);
        bval = m;
    }
    __syncthreads();
    const float bmax = bval;

    float s = 0.f;
    for (int j = tid; j < TP; j += 256) {
        float e = 0.f;
        if (j < Tn) { e = __expf(row[j] * scale + br[j] - bmax); s += e; }
        row[j] = e;
    }
    __syncthreads();
    #pragma unroll
    for (int o = 16; o; o >>= 1) s += __shfl_xor(s, o, 32);
    if ((tid & 31) == 0) red[tid >> 5] = s;
    __syncthreads();
    if (tid == 0) {
        float m = 0.f;
        for (int k = 0; k < 8; ++k) m += red[k];
        bval = m;
    }
    __syncthreads();
    const float inv = 1.0f / bval;
    for (int j = tid; j < Tn; j += 256) row[j] *= inv;
}

// ---------------------------------------------------------------------------
// gather padded x (B,TP,D) -> out (B,T,D)
// ---------------------------------------------------------------------------
__global__ void out_ker(const float* __restrict__ x, float* __restrict__ out)
{
    const long long idx = (long long)blockIdx.x * 256 + threadIdx.x;
    const long long total = (long long)Bn * Tn * Dn;
    if (idx >= total) return;
    const int d = (int)(idx % Dn);
    const long long btt = idx / Dn;
    const int b = (int)(btt / Tn);
    const int t = (int)(btt % Tn);
    out[idx] = x[((long long)b * TP + t) * Dn + d];
}

// ---------------------------------------------------------------------------
extern "C" void kernel_launch(void* const* d_in, const int* in_sizes, int n_in,
                              void* d_out, int out_size, void* d_ws, size_t ws_size,
                              hipStream_t stream)
{
    (void)in_sizes; (void)n_in; (void)out_size; (void)ws_size;
    const float* node_feats = (const float*)d_in[0];
    const int*   adj        = (const int*)  d_in[1];
    const float* dist       = (const float*)d_in[2];
    const float* eoh        = (const float*)d_in[3];
    const float* npw        = (const float*)d_in[4];
    const float* npb        = (const float*)d_in[5];
    const float* cls        = (const float*)d_in[6];
    const float* etw        = (const float*)d_in[7];
    const float* p_noedge   = (const float*)d_in[8];
    const float* p_degs     = (const float*)d_in[9];
    const float* p_gamma    = (const float*)d_in[10];
    const float* p_v2n      = (const float*)d_in[11];
    const float* p_n2v      = (const float*)d_in[12];
    const float* p_vself    = (const float*)d_in[13];
    const float* ln1g       = (const float*)d_in[14];
    const float* ln1b       = (const float*)d_in[15];
    const float* qw         = (const float*)d_in[16];
    const float* qbv        = (const float*)d_in[17];
    const float* kw         = (const float*)d_in[18];
    const float* kbv        = (const float*)d_in[19];
    const float* vw         = (const float*)d_in[20];
    const float* vbv        = (const float*)d_in[21];
    const float* ow         = (const float*)d_in[22];
    const float* obv        = (const float*)d_in[23];
    const float* ln2g       = (const float*)d_in[24];
    const float* ln2b       = (const float*)d_in[25];
    const float* fc1w       = (const float*)d_in[26];
    const float* fc1b       = (const float*)d_in[27];
    const float* fc2w       = (const float*)d_in[28];
    const float* fc2b       = (const float*)d_in[29];

    float* ws = (float*)d_ws;
    const long long XSZ = (long long)MP * Dn;       // 1,703,936 floats
    float* x    = ws;  ws += XSZ;
    float* xn   = ws;  ws += XSZ;
    float* qb_  = ws;  ws += XSZ;
    float* kb_  = ws;  ws += XSZ;
    float* vb_  = ws;  ws += XSZ;
    float* ao   = ws;  ws += XSZ;
    float* m1   = ws;  ws += XSZ;
    float* bias = ws;  ws += (long long)Bn * Tn * Tn;
    float* deg  = ws;  ws += Bn * Nn;
    float* dmx  = ws;  ws += Bn;
    float* sc   = ws;  ws += (long long)Hn * TP * TP;  // reused per (layer, b)

    // ---- bias construction -------------------------------------------------
    deg_ker   <<<Bn * Nn / 8, 256, 0, stream>>>(adj, deg);
    degmax_ker<<<Bn,          256, 0, stream>>>(deg, dmx);
    bias_ker  <<<Bn * Tn,     256, 0, stream>>>(adj, dist, eoh, etw, p_noedge, p_degs,
                                                p_gamma, p_v2n, p_n2v, p_vself, deg, dmx, bias);
    // ---- x = [cls ; node_feats @ npw + npb], padded ------------------------
    init_ker<<<Bn * 64, 256, 0, stream>>>(cls, x);
    for (int b = 0; b < Bn; ++b) {
        gemm_ker<64, false, true, false, false><<<dim3(Dn / 64, Nn / 64, 1), 256, 0, stream>>>(
            node_feats + (long long)b * Nn * Fn, Fn, 0,
            npw, Dn, 0, npb, nullptr,
            x + ((long long)b * TP + 1) * Dn, Dn, 0, Fn);
    }
    // ---- transformer layers ------------------------------------------------
    for (int l = 0; l < Ln; ++l) {
        ln_ker<<<Bn * Tn, 256, 0, stream>>>(x, ln1g + l * Dn, ln1b + l * Dn, xn);
        gemm_ker<64, false, true, false, false><<<dim3(Dn / 64, MP / 64, 1), 256, 0, stream>>>(
            xn, Dn, 0, qw + (long long)l * Dn * Dn, Dn, 0, qbv + l * Dn, nullptr, qb_, Dn, 0, Dn);
        gemm_ker<64, false, true, false, false><<<dim3(Dn / 64, MP / 64, 1), 256, 0, stream>>>(
            xn, Dn, 0, kw + (long long)l * Dn * Dn, Dn, 0, kbv + l * Dn, nullptr, kb_, Dn, 0, Dn);
        gemm_ker<64, false, true, false, false><<<dim3(Dn / 64, MP / 64, 1), 256, 0, stream>>>(
            xn, Dn, 0, vw + (long long)l * Dn * Dn, Dn, 0, vbv + l * Dn, nullptr, vb_, Dn, 0, Dn);

        for (int b = 0; b < Bn; ++b) {
            // scores[h,i,j] = Q_h[i,:] . K_h[j,:]   (B^T mode, z = head)
            gemm_ker<64, true, false, false, false><<<dim3(TP / 64, TP / 64, Hn), 256, 0, stream>>>(
                qb_ + (long long)b * TP * Dn, Dn, DKn,
                kb_ + (long long)b * TP * Dn, Dn, DKn,
                nullptr, nullptr, sc, TP, (long long)TP * TP, DKn);
            softmax_ker<<<dim3(Tn, Hn), 256, 0, stream>>>(sc, bias, b);
            // out_h = P_h @ V_h  (N = 32 -> narrow tile)
            gemm_ker<32, false, false, false, false><<<dim3(1, TP / 64, Hn), 256, 0, stream>>>(
                sc, TP, (long long)TP * TP,
                vb_ + (long long)b * TP * Dn, Dn, DKn,
                nullptr, nullptr, ao + (long long)b * TP * Dn, Dn, DKn, TP);
        }
        // x += ao @ ow + ob
        gemm_ker<64, false, true, false, true><<<dim3(Dn / 64, MP / 64, 1), 256, 0, stream>>>(
            ao, Dn, 0, ow + (long long)l * Dn * Dn, Dn, 0, obv + l * Dn, x, x, Dn, 0, Dn);
        // MLP
        ln_ker<<<Bn * Tn, 256, 0, stream>>>(x, ln2g + l * Dn, ln2b + l * Dn, xn);
        gemm_ker<64, false, true, true, false><<<dim3(Dm / 64, MP / 64, 1), 256, 0, stream>>>(
            xn, Dn, 0, fc1w + (long long)l * Dn * Dm, Dm, 0, fc1b + l * Dm, nullptr, m1, Dm, 0, Dn);
        gemm_ker<64, false, true, false, true><<<dim3(Dn / 64, MP / 64, 1), 256, 0, stream>>>(
            m1, Dm, 0, fc2w + (long long)l * Dm * Dn, Dn, 0, fc2b + l * Dn, x, x, Dn, 0, Dm);
    }
    out_ker<<<(Bn * Tn * Dn) / 256, 256, 0, stream>>>(x, (float*)d_out);
}
